// SamplerQNNGen_65481071399326
// MI455X (gfx1250) — matrix operand order repaired and checked by reading, not compiled, not run on hardware
//
#include <hip/hip_runtime.h>
#include <hip/hip_bf16.h>

typedef __attribute__((ext_vector_type(16))) _Float16 v16h;
typedef __attribute__((ext_vector_type(8)))  float    v8f;
typedef __attribute__((ext_vector_type(2)))  _Float16 h2;

#define WAVES_PER_BLOCK 8
#define THREADS (WAVES_PER_BLOCK * 32)

union HV { v16h v; h2 p[8]; };

// gfx1250 has a hardware V_TANH_F32 transcendental
__device__ __forceinline__ float fast_tanh(float v) {
#if __has_builtin(__builtin_amdgcn_tanhf)
  return __builtin_amdgcn_tanhf(v);
#else
  return tanhf(v);
#endif
}

__device__ __forceinline__ float fast_rcp(float v) {
  return __builtin_amdgcn_rcpf(v);
}

__global__ __launch_bounds__(THREADS, 1)
void qnn_fused_wmma_kernel(
    const float* __restrict__ x,
    const float* __restrict__ conv_w, const float* __restrict__ conv_b,
    const float* __restrict__ e0_w,  const float* __restrict__ e0_b,
    const float* __restrict__ e1_w,  const float* __restrict__ e1_b,
    const float* __restrict__ e2_w,  const float* __restrict__ e2_b,
    const float* __restrict__ fm_w,  const float* __restrict__ fm_b,
    const float* __restrict__ c1_w,  const float* __restrict__ c1_b,
    const float* __restrict__ p1_w,  const float* __restrict__ p1_b,
    const float* __restrict__ c2_w,  const float* __restrict__ c2_b,
    const float* __restrict__ p2_w,  const float* __restrict__ p2_b,
    const float* __restrict__ c3_w,  const float* __restrict__ c3_b,
    const float* __restrict__ head_w, const float* __restrict__ head_b,
    float* __restrict__ out, int batch)
{
  __shared__ float sLat[WAVES_PER_BLOCK][16][2];

  const int lane = threadIdx.x & 31;
  const int wid  = threadIdx.x >> 5;
  const int hf   = lane >> 4;   // which 16-lane half
  const int l    = lane & 15;

  // conv weights (wave-uniform -> SGPRs)
  const float cw0 = conv_w[0], cw1 = conv_w[1], cw2 = conv_w[2], cw3 = conv_w[3];
  const float cb  = conv_b[0];

  // e0 weight/bias f16 pairs in registers, arranged to match the 16-bit
  // A-matrix layout: lane half 0 owns K = {0..7,16..23}+32c, half 1 owns
  // K = {8..15,24..31}+32c, packed as (k, k+1) pairs per VGPR.
  h2 w0p[32], b0p[32];
  #pragma unroll
  for (int c = 0; c < 4; ++c)
  #pragma unroll
  for (int vg = 0; vg < 8; ++vg) {
    int j  = c * 8 + vg;
    int kb = 32 * c + ((vg >> 2) << 4) + (hf << 3) + ((vg & 3) << 1);
    h2 w; w.x = (_Float16)e0_w[kb]; w.y = (_Float16)e0_w[kb + 1]; w0p[j] = w;
    h2 b; b.x = (_Float16)e0_b[kb]; b.y = (_Float16)e0_b[kb + 1]; b0p[j] = b;
  }

  // e1 weights as f16 WMMA B-matrices (32x16 per K-chunk): lane = column n,
  // half 0 holds K=kc+{0..7,16..23}, half 1 holds K=kc+{8..15,24..31}, packed pairs per VGPR.
  HV Bm[4][4];
  #pragma unroll
  for (int t = 0; t < 4; ++t)
  #pragma unroll
  for (int c = 0; c < 4; ++c)
  #pragma unroll
  for (int vg = 0; vg < 8; ++vg) {
    int n = 16 * t + l;
    int k = 32 * c + ((vg >> 2) << 4) + (hf << 3) + ((vg & 3) << 1);
    h2 w; w.x = (_Float16)e1_w[n * 128 + k]; w.y = (_Float16)e1_w[n * 128 + k + 1];
    Bm[t][c].p[vg] = w;
  }

  // e1 bias + e2 rows for this lane's 4 output columns
  float b1v[4], w2a[4], w2b[4];
  #pragma unroll
  for (int t = 0; t < 4; ++t) {
    int n = 16 * t + l;
    b1v[t] = e1_b[n]; w2a[t] = e2_w[n]; w2b[t] = e2_w[64 + n];
  }
  const float e2b0 = e2_b[0], e2b1 = e2_b[1];

  const int tilesTotal = batch >> 4;
  const int waveGlobal = blockIdx.x * WAVES_PER_BLOCK + wid;
  const int waveCount  = gridDim.x * WAVES_PER_BLOCK;
  const float4* xv = (const float4*)x;
  float2* outv = (float2*)out;

  for (int tile = waveGlobal; tile < tilesTotal; tile += waveCount) {
    const int m0 = tile << 4;

    // conv + sigmoid for row m0+l (computed redundantly on both halves,
    // matching the A-matrix M mapping). sigmoid(z) = 0.5*tanh(z/2)+0.5:
    // one hardware TRANS op instead of exp + divide.
    float4 xe = xv[m0 + l];
    float z = fmaf(xe.x, cw0, fmaf(xe.y, cw1, fmaf(xe.z, cw2, fmaf(xe.w, cw3, cb))));
    float s = fmaf(0.5f, fast_tanh(0.5f * z), 0.5f);
    h2 s2; s2.x = (_Float16)s; s2.y = s2.x;

    // A = relu(s*w0 + b0), built directly in WMMA A layout via packed f16 math
    HV a[4];
    #pragma unroll
    for (int c = 0; c < 4; ++c)
    #pragma unroll
    for (int vg = 0; vg < 8; ++vg) {
      int j = c * 8 + vg;
      h2 hv = s2 * w0p[j] + b0p[j];
      hv.x = hv.x > (_Float16)0 ? hv.x : (_Float16)0;
      hv.y = hv.y > (_Float16)0 ? hv.y : (_Float16)0;
      a[c].p[vg] = hv;
    }

    float lat0[8], lat1[8];
    #pragma unroll
    for (int r = 0; r < 8; ++r) { lat0[r] = 0.f; lat1[r] = 0.f; }

    // e1 GEMM: 16x64 = sum over 4 K-chunks, fused relu + e2 projection
    #pragma unroll
    for (int t = 0; t < 4; ++t) {
      v8f acc = {0.f, 0.f, 0.f, 0.f, 0.f, 0.f, 0.f, 0.f};
      #pragma unroll
      for (int c = 0; c < 4; ++c)
        acc = __builtin_amdgcn_wmma_f32_16x16x32_f16(
            false, a[c].v, false, Bm[t][c].v, (short)0, acc, false, false);
      #pragma unroll
      for (int r = 0; r < 8; ++r) {
        float g = fmaxf(acc[r] + b1v[t], 0.0f);   // n = 16t+l, m = r+8*hf
        lat0[r] = fmaf(g, w2a[t], lat0[r]);
        lat1[r] = fmaf(g, w2b[t], lat1[r]);
      }
    }

    // reduce over n (16 lanes of each half; xor<16 never crosses halves)
    #pragma unroll
    for (int r = 0; r < 8; ++r) {
      #pragma unroll
      for (int off = 8; off >= 1; off >>= 1) {
        lat0[r] += __shfl_xor(lat0[r], off, 32);
        lat1[r] += __shfl_xor(lat1[r], off, 32);
      }
    }

    // per-wave LDS transpose: lane0 -> rows 0..7, lane16 -> rows 8..15
    if (l == 0) {
      #pragma unroll
      for (int r = 0; r < 8; ++r) {
        sLat[wid][(hf << 3) + r][0] = lat0[r];
        sLat[wid][(hf << 3) + r][1] = lat1[r];
      }
    }
    // same-wave DS ops are in-order; compiler inserts s_wait_dscnt
    if (lane < 16) {
      float L0 = sLat[wid][lane][0] + e2b0;
      float L1 = sLat[wid][lane][1] + e2b1;
      // fm: 2->4
      float y0[4];
      #pragma unroll
      for (int i = 0; i < 4; ++i)
        y0[i] = fast_tanh(fmaf(fm_w[2*i], L0, fmaf(fm_w[2*i+1], L1, fm_b[i])));
      // c1: 4->4
      float y1[4];
      #pragma unroll
      for (int i = 0; i < 4; ++i) {
        float v = c1_b[i];
        #pragma unroll
        for (int j = 0; j < 4; ++j) v = fmaf(c1_w[4*i+j], y0[j], v);
        y1[i] = fast_tanh(v);
      }
      // p1: 4->3
      float y2[3];
      #pragma unroll
      for (int i = 0; i < 3; ++i) {
        float v = p1_b[i];
        #pragma unroll
        for (int j = 0; j < 4; ++j) v = fmaf(p1_w[4*i+j], y1[j], v);
        y2[i] = fast_tanh(v);
      }
      // c2: 3->2
      float y3[2];
      #pragma unroll
      for (int i = 0; i < 2; ++i) {
        float v = c2_b[i];
        #pragma unroll
        for (int j = 0; j < 3; ++j) v = fmaf(c2_w[3*i+j], y2[j], v);
        y3[i] = fast_tanh(v);
      }
      // p2: 2->1, c3: 1->1, head: 1->2
      float y4 = fast_tanh(fmaf(p2_w[0], y3[0], fmaf(p2_w[1], y3[1], p2_b[0])));
      float y5 = fast_tanh(fmaf(c3_w[0], y4, c3_b[0]));
      float z0 = fmaf(head_w[0], y5, head_b[0]);
      float z1 = fmaf(head_w[1], y5, head_b[1]);
      // softmax over 2
      float mx = fmaxf(z0, z1);
      float ea = __expf(z0 - mx), eb = __expf(z1 - mx);
      float inv = fast_rcp(ea + eb);
      outv[m0 + lane] = make_float2(ea * inv, eb * inv);
    }
  }
}

extern "C" void kernel_launch(void* const* d_in, const int* in_sizes, int n_in,
                              void* d_out, int out_size, void* d_ws, size_t ws_size,
                              hipStream_t stream) {
  (void)n_in; (void)out_size; (void)d_ws; (void)ws_size;
  const float* x      = (const float*)d_in[0];
  const float* conv_w = (const float*)d_in[1];
  const float* conv_b = (const float*)d_in[2];
  const float* e0_w   = (const float*)d_in[3];
  const float* e0_b   = (const float*)d_in[4];
  const float* e1_w   = (const float*)d_in[5];
  const float* e1_b   = (const float*)d_in[6];
  const float* e2_w   = (const float*)d_in[7];
  const float* e2_b   = (const float*)d_in[8];
  const float* fm_w   = (const float*)d_in[9];
  const float* fm_b   = (const float*)d_in[10];
  const float* c1_w   = (const float*)d_in[11];
  const float* c1_b   = (const float*)d_in[12];
  const float* p1_w   = (const float*)d_in[13];
  const float* p1_b   = (const float*)d_in[14];
  const float* c2_w   = (const float*)d_in[15];
  const float* c2_b   = (const float*)d_in[16];
  const float* p2_w   = (const float*)d_in[17];
  const float* p2_b   = (const float*)d_in[18];
  const float* c3_w   = (const float*)d_in[19];
  const float* c3_b   = (const float*)d_in[20];
  const float* head_w = (const float*)d_in[21];
  const float* head_b = (const float*)d_in[22];

  int batch = in_sizes[0] / 4;   // x is (B,1,2,2)

  dim3 grid(1024), block(THREADS);
  qnn_fused_wmma_kernel<<<grid, block, 0, stream>>>(
      x, conv_w, conv_b, e0_w, e0_b, e1_w, e1_b, e2_w, e2_b,
      fm_w, fm_b, c1_w, c1_b, p1_w, p1_b, c2_w, c2_b,
      p2_w, p2_b, c3_w, c3_b, head_w, head_b,
      (float*)d_out, batch);
}
